// Sequence_26860725469223
// MI455X (gfx1250) — matrix operand order repaired and checked by reading, not compiled, not run on hardware
//
#include <hip/hip_runtime.h>

typedef __attribute__((ext_vector_type(16))) _Float16 v16h;
typedef __attribute__((ext_vector_type(8)))  _Float16 v8h;
typedef __attribute__((ext_vector_type(8)))  float    v8f;

#define T_IN   2048
#define FUT    64
#define TOUT   (T_IN + FUT)
#define NFRAG  24                    // 8 layer-1 B frags + 16 layer-2 B frags
#define WGT_BYTES (NFRAG * 1024)     // 24576
#define H1OFF  WGT_BYTES
#define H2OFF  (H1OFF + 4096)        // 4 waves * 16 rows * 32 halves * 2B
#define OUTOFF (H2OFF + 4096)
#define SMEM_BYTES (OUTOFF + 4 * 16 * 4)

__device__ __forceinline__ float sig_(float x)  { return 1.0f / (1.0f + __expf(-x)); }
__device__ __forceinline__ float tanh_(float x) { return 1.0f - 2.0f / (1.0f + __expf(2.0f * x)); }

// load a 16-half fragment from two 16B-aligned LDS locations
__device__ __forceinline__ v16h ldfrag(const _Float16* lo, const _Float16* hi) {
  v8h a = *(const v8h*)lo;
  v8h b = *(const v8h*)hi;
  v16h r;
#pragma unroll
  for (int i = 0; i < 8; i++) { r[i] = a[i]; r[i + 8] = b[i]; }
  return r;
}

struct StepCtx {
  const unsigned char* wgt;
  _Float16* h1p;
  _Float16* h2p;
  float*    outp;
  float*    out;
  int rowbase, c, halfid, lane;
  float wl0, wl1, blin;
};

// One LSTM time step for a 16-row batch tile owned by one wave32.
__device__ __forceinline__ void lstm_step(const StepCtx& S, float* c1s, float* c2s,
                                          float xv, int t) {
  const int c = S.c, halfid = S.halfid, lane = S.lane;
  const v8f zf = {};

  // ================= layer 1: gates1 = [h1 | x | 1] @ W1p^T =================
  const _Float16* arow = S.h1p + c * 32;               // A row = c
  v16h a1 = ldfrag(arow + 8 * halfid, arow + 16 + 8 * halfid);
  if (halfid) {                                        // lanes 16-31 hold K=24..31
    a1[9]  = (_Float16)xv;                             // K = 25 : x_t
    a1[10] = (_Float16)1.0f;                           // K = 26 : bias enable
  }
  v8f acc[8];
#pragma unroll
  for (int n = 0; n < 8; n++) {
    const _Float16* bp = (const _Float16*)(S.wgt + n * 1024 + lane * 32);
    v16h bf = ldfrag(bp, bp + 8);
    acc[n] = __builtin_amdgcn_wmma_f32_16x16x32_f16(false, a1, false, bf,
                                                    (short)0, zf, false, false);
  }
  // elementwise cell, in C-fragment layout (row = r+8*halfid, col = 16*t2+c)
  float hh1[2][8];
#pragma unroll
  for (int t2 = 0; t2 < 2; t2++) {
#pragma unroll
    for (int r = 0; r < 8; r++) {
      int ci = t2 * 8 + r;
      float iv = sig_(acc[0 + t2][r]);
      float fv = sig_(acc[2 + t2][r]);
      float gv = tanh_(acc[4 + t2][r]);
      float ov = sig_(acc[6 + t2][r]);
      float cn = fv * c1s[ci] + iv * gv;
      c1s[ci] = cn;
      hh1[t2][r] = ov * tanh_(cn);
    }
  }
  // stage h1_new to LDS: col 25 <- 1.0 (layer-2 bias slot), col>25 <- 0
#pragma unroll
  for (int t2 = 0; t2 < 2; t2++) {
    int col = 16 * t2 + c;
#pragma unroll
    for (int r = 0; r < 8; r++) {
      float v = hh1[t2][r];
      if (col == 25)     v = 1.0f;
      else if (col > 25) v = 0.0f;
      S.h1p[(r + 8 * halfid) * 32 + col] = (_Float16)v;
    }
  }

  // ============ layer 2: gates2 = [h1,1 | h2] @ [W_ih2 ; W_hh2]^T ============
  v16h a20 = ldfrag(arow + 8 * halfid, arow + 16 + 8 * halfid);
  const _Float16* arow2 = S.h2p + c * 32;
  v16h a21 = ldfrag(arow2 + 8 * halfid, arow2 + 16 + 8 * halfid);
#pragma unroll
  for (int n = 0; n < 8; n++) {
    const _Float16* b0 = (const _Float16*)(S.wgt + (8 + 2 * n) * 1024 + lane * 32);
    const _Float16* b1 = (const _Float16*)(S.wgt + (9 + 2 * n) * 1024 + lane * 32);
    v16h bf0 = ldfrag(b0, b0 + 8);
    v8f  ta  = __builtin_amdgcn_wmma_f32_16x16x32_f16(false, a20, false, bf0,
                                                      (short)0, zf, false, false);
    v16h bf1 = ldfrag(b1, b1 + 8);
    acc[n] = __builtin_amdgcn_wmma_f32_16x16x32_f16(false, a21, false, bf1,
                                                    (short)0, ta, false, false);
  }
  float hh2[2][8];
#pragma unroll
  for (int t2 = 0; t2 < 2; t2++) {
#pragma unroll
    for (int r = 0; r < 8; r++) {
      int ci = t2 * 8 + r;
      float iv = sig_(acc[0 + t2][r]);
      float fv = sig_(acc[2 + t2][r]);
      float gv = tanh_(acc[4 + t2][r]);
      float ov = sig_(acc[6 + t2][r]);
      float cn = fv * c2s[ci] + iv * gv;
      c2s[ci] = cn;
      hh2[t2][r] = ov * tanh_(cn);
    }
  }
#pragma unroll
  for (int t2 = 0; t2 < 2; t2++) {
    int col = 16 * t2 + c;
#pragma unroll
    for (int r = 0; r < 8; r++) {
      float v = (col < 25) ? hh2[t2][r] : 0.0f;
      S.h2p[(r + 8 * halfid) * 32 + col] = (_Float16)v;
    }
  }

  // ================= out = h2 . W_lin + b (and feedback for future) =========
  float s[8];
#pragma unroll
  for (int r = 0; r < 8; r++) {
    float pr = hh2[0][r] * S.wl0 + hh2[1][r] * S.wl1;  // wl1==0 for cols>=25
#pragma unroll
    for (int off = 1; off < 16; off <<= 1)
      pr += __shfl_xor(pr, off, 32);                   // reduce within 16-lane half
    s[r] = pr + S.blin;
  }
  float vout = s[0];
  if (c == 1) vout = s[1];
  if (c == 2) vout = s[2];
  if (c == 3) vout = s[3];
  if (c == 4) vout = s[4];
  if (c == 5) vout = s[5];
  if (c == 6) vout = s[6];
  if (c == 7) vout = s[7];
  if (c < 8) {
    int row = c + 8 * halfid;
    S.out[(size_t)(S.rowbase + row) * TOUT + t] = vout;
    S.outp[row] = vout;                                // next-step x for future phase
  }
}

__global__ __launch_bounds__(128) void lstm2_persistent(
    const float* __restrict__ input,
    const float* __restrict__ W_ih1, const float* __restrict__ W_hh1,
    const float* __restrict__ b_ih1, const float* __restrict__ b_hh1,
    const float* __restrict__ W_ih2, const float* __restrict__ W_hh2,
    const float* __restrict__ b_ih2, const float* __restrict__ b_hh2,
    const float* __restrict__ W_lin, const float* __restrict__ b_lin,
    float* __restrict__ out)
{
  __shared__ __align__(16) unsigned char smem[SMEM_BYTES];
  const int tid    = threadIdx.x;
  const int lane   = tid & 31;
  const int wave   = tid >> 5;
  const int c      = lane & 15;     // N column / A row within tile
  const int halfid = lane >> 4;

  // ---- one-time: swizzle padded weights into B-fragment layout in LDS ----
  // Fragment f, lane l holds 16 halves (K = 8*(l>=16) + e + 8*(e>=8), col = l&15) at
  // smem[f*1024 + l*32 + e*2].  Gate-padded columns: p in [0,128), gate G=p/32,
  // within-gate q=p%32 (valid q<25 -> weight row 25*G+q).
  for (int idx = tid; idx < NFRAG * 512; idx += 128) {
    int f   = idx >> 9;
    int rem = idx & 511;
    int l   = rem >> 4;
    int e   = rem & 15;
    int K   = (e < 8) ? (((l >= 16) ? 8 : 0) + e) : (16 + ((l >= 16) ? 8 : 0) + (e - 8));
    int cc  = l & 15;
    int n, kt, layer;
    if (f < 8) { n = f;            kt = 0;            layer = 1; }
    else       { n = (f - 8) >> 1; kt = (f - 8) & 1;  layer = 2; }
    int p = 16 * n + cc;
    int G = p >> 5, q = p & 31;
    float val = 0.0f;
    if (q < 25) {
      int row = 25 * G + q;
      if (layer == 1) {
        if      (K < 25)  val = W_hh1[row * 25 + K];
        else if (K == 25) val = W_ih1[row];                      // x slot
        else if (K == 26) val = b_ih1[row] + b_hh1[row];         // bias slot
      } else if (kt == 0) {
        if      (K < 25)  val = W_ih2[row * 25 + K];
        else if (K == 25) val = b_ih2[row] + b_hh2[row];         // bias slot
      } else {
        if      (K < 25)  val = W_hh2[row * 25 + K];
      }
    }
    *((_Float16*)(smem + f * 1024 + l * 32 + e * 2)) = (_Float16)val;
  }
  // zero h1/h2/out staging buffers
  for (int b = tid * 4; b < 4096 + 4096 + 256; b += 128 * 4)
    *((float*)(smem + H1OFF + b)) = 0.0f;
  __syncthreads();

  StepCtx S;
  S.wgt     = smem;
  S.h1p     = (_Float16*)(smem + H1OFF + wave * 1024);
  S.h2p     = (_Float16*)(smem + H2OFF + wave * 1024);
  S.outp    = (float*)(smem + OUTOFF + wave * 64);
  S.out     = out;
  S.rowbase = blockIdx.x * 64 + wave * 16;
  S.c = c; S.halfid = halfid; S.lane = lane;
  S.wl0  = W_lin[c];
  S.wl1  = (16 + c < 25) ? W_lin[16 + c] : 0.0f;
  S.blin = b_lin[0];

  float c1s[16], c2s[16];
#pragma unroll
  for (int i = 0; i < 16; i++) { c1s[i] = 0.0f; c2s[i] = 0.0f; }

  // Phase 1: driven by the input sequence (pure global_load_b32 on a
  // strength-reduced pointer; no flat path, no per-step select).
  const float* xptr = input + (size_t)(S.rowbase + c) * T_IN;
#pragma clang loop unroll(disable)
  for (int t = 0; t < T_IN; t++) {
    float xv = xptr[t];
    lstm_step(S, c1s, c2s, xv, t);
  }

  // Phase 2: autoregressive — x is last step's output, read from LDS.
#pragma clang loop unroll(disable)
  for (int t = T_IN; t < TOUT; t++) {
    float xv = S.outp[c];
    lstm_step(S, c1s, c2s, xv, t);
  }
}

extern "C" void kernel_launch(void* const* d_in, const int* in_sizes, int n_in,
                              void* d_out, int out_size, void* d_ws, size_t ws_size,
                              hipStream_t stream) {
  (void)in_sizes; (void)n_in; (void)out_size; (void)d_ws; (void)ws_size;
  const float* input = (const float*)d_in[0];
  const float* W_ih1 = (const float*)d_in[1];
  const float* W_hh1 = (const float*)d_in[2];
  const float* b_ih1 = (const float*)d_in[3];
  const float* b_hh1 = (const float*)d_in[4];
  const float* W_ih2 = (const float*)d_in[5];
  const float* W_hh2 = (const float*)d_in[6];
  const float* b_ih2 = (const float*)d_in[7];
  const float* b_hh2 = (const float*)d_in[8];
  const float* W_lin = (const float*)d_in[9];
  const float* b_lin = (const float*)d_in[10];
  float* outp = (float*)d_out;

  lstm2_persistent<<<dim3(8), dim3(128), 0, stream>>>(
      input, W_ih1, W_hh1, b_ih1, b_hh1,
      W_ih2, W_hh2, b_ih2, b_hh2, W_lin, b_lin, outp);
}
